// Autoformer_48000554500801
// MI455X (gfx1250) — matrix operand order repaired and checked by reading, not compiled, not run on hardware
//
#include <hip/hip_runtime.h>
#include <hip/hip_bf16.h>
#include <math.h>

// ---------------------------------------------------------------------------
// Autoformer encoder classifier for MI455X (gfx1250, wave32, WMMA + TDM).
//  - All GEMM operands pre-converted to bf16 (once per producer, fused or
//    via tiny bandwidth passes); GEMM inner loop is ds_load_b128 + v_wmma
//    only (no per-step conversion VALU).
//  - TDM (tensor_load_to_lds, 6-arg clang-23 builtin) moves bf16 tiles,
//    double-buffered, TENSORcnt-synced; TDM pad => 66-elem LDS row stride.
//  - BK=64: 8 WMMAs per wave per barrier, K/64 DMA steps.
//  - FFT autocorrelation == circular diagonals of Q_b K_b^T (WMMA +
//    ds_add_f32 bins + global_atomic_add flush).
// ---------------------------------------------------------------------------

#define B_      32
#define L_      1024
#define DM_     256
#define DFF_    1024
#define CIN_    21
#define NL_     3
#define TOPK_   6

#define BM_     128          // block tile M
#define BN_     64           // block tile N
#define BKE_    64           // K elements per LDS tile step (2 WMMA k-chunks)
#define LDSTE_  66           // LDS row stride in bf16 elems (64 + 1 pad dword)

typedef __attribute__((ext_vector_type(16))) __bf16        v16bf;
typedef __attribute__((ext_vector_type(8)))  float         v8f;
typedef __attribute__((ext_vector_type(8)))  int           v8i;
typedef __attribute__((ext_vector_type(4)))  unsigned int  u32x4;
typedef __attribute__((ext_vector_type(8)))  int           i32x8;
typedef __attribute__((ext_vector_type(4)))  int           i32x4;

typedef unsigned short ushort_t;

__device__ __forceinline__ ushort_t f2bf(float f) {
  unsigned u = __float_as_uint(f);
  unsigned r = u + 0x7FFFu + ((u >> 16) & 1u);   // round-to-nearest-even
  return (ushort_t)(r >> 16);
}

__device__ __forceinline__ float gelu_exact(float v) {
  return 0.5f * v * (1.0f + erff(v * 0.70710678118654752f));
}

// ---------------------------------------------------------------------------
// TDM: 2D bf16 tile (tile1 rows x 64 cols) global -> LDS, double-buffer src.
// D# per CDNA5 ISA 8.3/8.4: data_size=2B, pad_enable, pad_interval=32 dwords
// (=64 bf16 = one row), pad_amount=1 dword (=2 elems) => row stride 66 elems.
// ---------------------------------------------------------------------------
__device__ __forceinline__ void tdm_load_2d(const ushort_t* gptr, unsigned ldsOff,
                                            int tile1, int K)
{
  unsigned long long ga = (unsigned long long)(uintptr_t)gptr;
  u32x4 g0;
  g0[0] = 1u;                                            // count=1, user desc
  g0[1] = ldsOff;                                        // lds_addr
  g0[2] = (unsigned)(ga & 0xFFFFFFFFu);                  // global_addr[31:0]
  g0[3] = (unsigned)((ga >> 32) & 0x01FFFFFFu) | (2u << 30); // addr[56:32],type=2
  i32x8 g1;
  g1[0] = (int)((1u << 16) | (1u << 20) | (4u << 22));   // 2B, pad_en, ivl=32dw, amt=1dw
  g1[1] = (int)((unsigned)(K & 0xFFFF) << 16);           // tensor_dim0 lo16
  g1[2] = (int)(((unsigned)K >> 16) | (0xFFFFu << 16));  // dim0 hi16 | dim1 lo16
  g1[3] = (int)(0x7FFFu | ((unsigned)BKE_ << 16));       // dim1 hi16 | tile_dim0=64
  g1[4] = (int)(unsigned)tile1;                          // tile_dim1 | tile_dim2=0
  g1[5] = (int)(unsigned)K;                              // tensor_dim0_stride lo32
  g1[6] = 0;
  g1[7] = 0;
  i32x4 z4 = {};
  i32x8 z8 = {};
  __builtin_amdgcn_tensor_load_to_lds(g0, g1, z4, z4, z8, 0);
}

// ---------------------------------------------------------------------------
// GEMM core: acc[2][2] (+)= A[BM_,K](bf16) x W[BN_,K](bf16)^T.
// 256 threads, 8 waves 4(M)x2(N); wave tile 32x32; 8 WMMA per step (BK=64).
// ---------------------------------------------------------------------------
__device__ __forceinline__ void gemm_tile_core(
    const ushort_t* __restrict__ A, const ushort_t* __restrict__ W, int K,
    int m0, int n0, ushort_t* sA, ushort_t* sB, v8f acc[2][2])
{
  const int tid  = threadIdx.x;
  const int wave = tid >> 5;
  const int lane = tid & 31;
  const int wm = (wave & 3) * 32;
  const int wn = (wave >> 2) * 32;

  // CDNA5 wave32 fragment coordinates (ISA 7.12.2)
  const int rowA = lane & 15;
  const int kba  = (lane >> 4) * 8;
  const int kbb  = (lane >> 4) * 16;

  const unsigned ldsA0 = (unsigned)(uintptr_t)sA;   // low 32 bits == LDS offset
  const unsigned ldsB0 = (unsigned)(uintptr_t)sB;

  if (wave == 0) {   // prime buffer 0
    tdm_load_2d(A + (size_t)m0 * K, ldsA0, BM_, K);
    tdm_load_2d(W + (size_t)n0 * K, ldsB0, BN_, K);
  }

  const int nsteps = K / BKE_;
  for (int s = 0; s < nsteps; ++s) {
    const int buf = s & 1;
    if (wave == 0) __builtin_amdgcn_s_wait_tensorcnt(0);
    __syncthreads();                      // tile `buf` visible to all waves

    if (wave == 0 && s + 1 < nsteps) {    // async-fill the other buffer
      const int nb = buf ^ 1;
      tdm_load_2d(A + (size_t)m0 * K + (s + 1) * BKE_,
                  ldsA0 + (unsigned)(nb * BM_ * LDSTE_ * 2), BM_, K);
      tdm_load_2d(W + (size_t)n0 * K + (s + 1) * BKE_,
                  ldsB0 + (unsigned)(nb * BN_ * LDSTE_ * 2), BN_, K);
    }

    const ushort_t* a0 = sA + buf * BM_ * LDSTE_;
    const ushort_t* b0 = sB + buf * BN_ * LDSTE_;

#pragma unroll
    for (int kc = 0; kc < 2; ++kc) {      // two K=32 WMMA chunks per tile
      v16bf af[2], bfr[2];
#pragma unroll
      for (int mf = 0; mf < 2; ++mf) {
        const ushort_t* ar = a0 + (wm + mf * 16 + rowA) * LDSTE_ + kc * 32;
        v8i r;
#pragma unroll
        for (int j = 0; j < 8; ++j) {
          int kl = (j < 4) ? (kba + 2 * j) : (16 + kba + 2 * (j - 4));
          r[j] = *(const int*)(ar + kl);
        }
        af[mf] = __builtin_bit_cast(v16bf, r);
      }
#pragma unroll
      for (int nf = 0; nf < 2; ++nf) {
        const ushort_t* br =
            b0 + (wn + nf * 16 + (lane & 15)) * LDSTE_ + kc * 32 + kba;
        v8i r;
#pragma unroll
        for (int j = 0; j < 8; ++j) r[j] = *(const int*)(br + ((j < 4) ? 2 * j : 16 + 2 * (j - 4)));
        bfr[nf] = __builtin_bit_cast(v16bf, r);
      }
#pragma unroll
      for (int mi = 0; mi < 2; ++mi)
#pragma unroll
        for (int ni = 0; ni < 2; ++ni)
          acc[mi][ni] = __builtin_amdgcn_wmma_f32_16x16x32_bf16(
              false, af[mi], false, bfr[ni], (short)0, acc[mi][ni], false, false);
    }
    // next TDM into `buf` is only issued after the *next* barrier -> safe.
  }
}

// ---------------------------------------------------------------------------
// C[M,N] = A[M,K] x W[N,K]^T (+bias) (+GELU); output fp32 or bf16.
// ---------------------------------------------------------------------------
template <bool GELU, bool OUTBF16>
__global__ __launch_bounds__(256)
void gemm_bf16_wmma(const ushort_t* __restrict__ A, const ushort_t* __restrict__ W,
                    const float* __restrict__ bias, void* __restrict__ Cv,
                    int N, int K)
{
  __shared__ __align__(16) ushort_t sA[2][BM_ * LDSTE_];
  __shared__ __align__(16) ushort_t sB[2][BN_ * LDSTE_];

  const int m0 = blockIdx.x * BM_;
  const int n0 = blockIdx.y * BN_;
  const int wave = threadIdx.x >> 5;
  const int lane = threadIdx.x & 31;
  const int wm = (wave & 3) * 32;
  const int wn = (wave >> 2) * 32;

  v8f acc[2][2] = {};
  gemm_tile_core(A, W, K, m0, n0, &sA[0][0], &sB[0][0], acc);

#pragma unroll
  for (int mi = 0; mi < 2; ++mi) {
    const int mloc = wm + mi * 16 + (lane >> 4) * 8;
#pragma unroll
    for (int ni = 0; ni < 2; ++ni) {
      const int gn = n0 + wn + ni * 16 + (lane & 15);
      const float bv = bias ? bias[gn] : 0.0f;
#pragma unroll
      for (int r = 0; r < 8; ++r) {
        float v = acc[mi][ni][r] + bv;
        if (GELU) v = gelu_exact(v);
        const size_t idx = (size_t)(m0 + mloc + r) * N + gn;
        if (OUTBF16) ((ushort_t*)Cv)[idx] = f2bf(v);
        else         ((float*)Cv)[idx]    = v;
      }
    }
  }
}

// ---------------------------------------------------------------------------
// corr[b,l] = (1/256) sum_t <q[b,(t+l)%L,:], k[b,t,:]>  via Gram diagonals.
// ---------------------------------------------------------------------------
__global__ __launch_bounds__(256)
void corr_wmma_kernel(const ushort_t* __restrict__ Q, const ushort_t* __restrict__ Km,
                      float* __restrict__ corr)
{
  __shared__ __align__(16) ushort_t sA[2][BM_ * LDSTE_];
  __shared__ __align__(16) ushort_t sB[2][BN_ * LDSTE_];
  __shared__ float diag[192];   // (m-n) - (m0-n0) in [-63, 127]

  const int b  = blockIdx.z;
  const int m0 = blockIdx.x * BM_;
  const int n0 = blockIdx.y * BN_;
  const int tid  = threadIdx.x;
  const int wave = tid >> 5;
  const int lane = tid & 31;
  const int wm = (wave & 3) * 32;
  const int wn = (wave >> 2) * 32;

  for (int i = tid; i < 192; i += 256) diag[i] = 0.0f;

  const ushort_t* A = Q  + (size_t)b * L_ * DM_;
  const ushort_t* W = Km + (size_t)b * L_ * DM_;

  v8f acc[2][2] = {};
  gemm_tile_core(A, W, DM_, m0, n0, &sA[0][0], &sB[0][0], acc);
  __syncthreads();

#pragma unroll
  for (int mi = 0; mi < 2; ++mi) {
    const int mloc = wm + mi * 16 + (lane >> 4) * 8;
#pragma unroll
    for (int ni = 0; ni < 2; ++ni) {
      const int nloc = wn + ni * 16 + (lane & 15);
#pragma unroll
      for (int r = 0; r < 8; ++r)
        atomicAdd(&diag[(mloc + r) - nloc + 63], acc[mi][ni][r]); // ds_add_f32
    }
  }
  __syncthreads();

  const int base = m0 - n0 - 63;
  for (int i = tid; i < 192; i += 256) {
    float v = diag[i];
    if (v != 0.0f)
      atomicAdd(&corr[(size_t)b * L_ + ((base + i) & (L_ - 1))],
                v * (1.0f / 256.0f));
  }
}

// fp32 -> bf16 bulk converter (weights etc.)
__global__ __launch_bounds__(256)
void cvt_bf16_kernel(const float* __restrict__ in, ushort_t* __restrict__ out, int n)
{
  int i = blockIdx.x * 256 + threadIdx.x;
  if (i < n) out[i] = f2bf(in[i]);
}

// ---------------------------------------------------------------------------
// Top-6 of corr[b,:] + softmax weights.  One block per batch.
// ---------------------------------------------------------------------------
__global__ __launch_bounds__(256)
void topk_softmax_kernel(const float* __restrict__ corr,
                         float* __restrict__ wv, int* __restrict__ dv)
{
  __shared__ float vals[L_];
  __shared__ float sv[256];
  __shared__ int   si[256];
  __shared__ float topw[TOPK_];
  __shared__ int   topd[TOPK_];
  const int b = blockIdx.x, t = threadIdx.x;

  for (int i = t; i < L_; i += 256) vals[i] = corr[(size_t)b * L_ + i];
  __syncthreads();

  for (int kk = 0; kk < TOPK_; ++kk) {
    float best = -1e30f; int bi = 0;
    for (int i = t; i < L_; i += 256)
      if (vals[i] > best) { best = vals[i]; bi = i; }
    sv[t] = best; si[t] = bi;
    __syncthreads();
    for (int off = 128; off > 0; off >>= 1) {
      if (t < off && sv[t + off] > sv[t]) { sv[t] = sv[t + off]; si[t] = si[t + off]; }
      __syncthreads();
    }
    if (t == 0) { topw[kk] = sv[0]; topd[kk] = si[0]; vals[si[0]] = -1e30f; }
    __syncthreads();
  }
  if (t == 0) {
    float m = topw[0];
    for (int k = 1; k < TOPK_; ++k) m = fmaxf(m, topw[k]);
    float e[TOPK_], s = 0.0f;
    for (int k = 0; k < TOPK_; ++k) { e[k] = __expf(topw[k] - m); s += e[k]; }
    float inv = 1.0f / s;
    for (int k = 0; k < TOPK_; ++k) {
      wv[b * TOPK_ + k] = e[k] * inv;
      dv[b * TOPK_ + k] = topd[k];
    }
  }
}

// out_bf16[b,l,c] = sum_k w[b,k] * v[b,(l+delay[b,k])%L, c]   (feeds O-proj)
__global__ __launch_bounds__(256)
void aggregate_kernel(const float* __restrict__ v, const float* __restrict__ wv,
                      const int* __restrict__ dv, ushort_t* __restrict__ out)
{
  const int c = threadIdx.x;
  const int b = blockIdx.x >> 10;
  const int l = blockIdx.x & (L_ - 1);
  const float* wb = wv + b * TOPK_;
  const int*   db = dv + b * TOPK_;
  const size_t base = (size_t)b * L_ * DM_;
  float s = 0.0f;
#pragma unroll
  for (int k = 0; k < TOPK_; ++k) {
    int idx = (l + db[k]) & (L_ - 1);
    s += wb[k] * v[base + (size_t)idx * DM_ + c];
  }
  out[base + (size_t)l * DM_ + c] = f2bf(s);
}

// out = s - movavg5(s), s = x + a; writes fp32 + bf16 copies.
__global__ __launch_bounds__(256)
void add_decomp_kernel(const float* __restrict__ x, const float* __restrict__ a,
                       float* __restrict__ out, ushort_t* __restrict__ outb)
{
  const int c = threadIdx.x;
  const int b = blockIdx.x >> 10;
  const int l = blockIdx.x & (L_ - 1);
  const size_t base = (size_t)b * L_ * DM_;
  float acc = 0.0f, center = 0.0f;
#pragma unroll
  for (int d = -2; d <= 2; ++d) {
    int ll = l + d; ll = ll < 0 ? 0 : (ll > L_ - 1 ? L_ - 1 : ll);
    size_t o = base + (size_t)ll * DM_ + c;
    float s = x[o] + a[o];
    acc += s;
    if (d == 0) center = s;
  }
  const float res = center - acc * 0.2f;
  const size_t o = base + (size_t)l * DM_ + c;
  out[o]  = res;
  outb[o] = f2bf(res);
}

// Token embedding: circular conv1d (kernel 3) C_IN=21 -> D_MODEL=256.
__global__ __launch_bounds__(256)
void token_embed_kernel(const float* __restrict__ xe, const float* __restrict__ tw,
                        float* __restrict__ out, ushort_t* __restrict__ outb)
{
  __shared__ float xs[3][CIN_];
  const int d = threadIdx.x;
  const int b = blockIdx.x >> 10;
  const int l = blockIdx.x & (L_ - 1);
  if (d < 3 * CIN_) {
    int j = d / CIN_, ci = d % CIN_;
    int ll = (l - 1 + j + L_) & (L_ - 1);          // wrap pad
    xs[j][ci] = xe[((size_t)b * L_ + ll) * CIN_ + ci];
  }
  __syncthreads();
  const float* w = tw + (size_t)d * (CIN_ * 3);
  float s = 0.0f;
#pragma unroll
  for (int ci = 0; ci < CIN_; ++ci)
#pragma unroll
    for (int j = 0; j < 3; ++j)
      s += xs[j][ci] * w[ci * 3 + j];
  const size_t o = ((size_t)b * L_ + l) * DM_ + d;
  out[o]  = s;
  outb[o] = f2bf(s);
}

// LayerNorm over last dim (256) with affine.  One block per (b,l).
__global__ __launch_bounds__(256)
void layernorm_kernel(const float* __restrict__ x, const float* __restrict__ w,
                      const float* __restrict__ bb, float* __restrict__ out)
{
  __shared__ float red[256];
  const int d = threadIdx.x;
  const size_t row = (size_t)blockIdx.x * DM_;
  float v = x[row + d];
  red[d] = v; __syncthreads();
  for (int off = 128; off > 0; off >>= 1) {
    if (d < off) red[d] += red[d + off];
    __syncthreads();
  }
  float mu = red[0] * (1.0f / DM_);
  __syncthreads();
  float c0 = v - mu;
  red[d] = c0 * c0; __syncthreads();
  for (int off = 128; off > 0; off >>= 1) {
    if (d < off) red[d] += red[d + off];
    __syncthreads();
  }
  float var = red[0] * (1.0f / DM_);
  out[row + d] = c0 * rsqrtf(var + 1e-5f) * w[d] + bb[d];
}

// Subtract per-(b,d) mean over L, then GELU.  Block per (d,b).
__global__ __launch_bounds__(256)
void colmean_gelu_kernel(const float* __restrict__ xh, float* __restrict__ out)
{
  __shared__ float red[256];
  const int t = threadIdx.x;
  const int d = blockIdx.x;
  const int b = blockIdx.y;
  const size_t base = (size_t)b * L_ * DM_ + d;
  float s = 0.0f;
#pragma unroll
  for (int i = 0; i < 4; ++i)
    s += xh[base + (size_t)(t + i * 256) * DM_];
  red[t] = s; __syncthreads();
  for (int off = 128; off > 0; off >>= 1) {
    if (t < off) red[t] += red[t + off];
    __syncthreads();
  }
  float mu = red[0] * (1.0f / L_);
#pragma unroll
  for (int i = 0; i < 4; ++i) {
    size_t o = base + (size_t)(t + i * 256) * DM_;
    out[o] = gelu_exact(xh[o] - mu);
  }
}

// Final projection: out[b,c] = <g[b,:], proj_w[c,:]> + proj_b[c]
__global__ __launch_bounds__(256)
void proj_kernel(const float* __restrict__ g, const float* __restrict__ pw,
                 const float* __restrict__ pb, float* __restrict__ out)
{
  __shared__ float red[256];
  const int t = threadIdx.x;
  const int c = blockIdx.x;
  const int b = blockIdx.y;
  const size_t NF = (size_t)L_ * DM_;   // 262144
  const float* gb  = g  + (size_t)b * NF;
  const float* pwc = pw + (size_t)c * NF;
  float s = 0.0f;
  for (size_t i = t; i < NF; i += 256) s += gb[i] * pwc[i];
  red[t] = s; __syncthreads();
  for (int off = 128; off > 0; off >>= 1) {
    if (t < off) red[t] += red[t + off];
    __syncthreads();
  }
  if (t == 0) out[b * 3 + c] = red[0] + pb[c];
}

// ---------------------------------------------------------------------------
extern "C" void kernel_launch(void* const* d_in, const int* in_sizes, int n_in,
                              void* d_out, int out_size, void* d_ws, size_t ws_size,
                              hipStream_t stream)
{
  (void)in_sizes; (void)n_in; (void)out_size; (void)ws_size;

  const float* x_enc   = (const float*)d_in[0];
  const float* token_w = (const float*)d_in[1];
  const float* qw = (const float*)d_in[2];
  const float* qb = (const float*)d_in[3];
  const float* kw = (const float*)d_in[4];
  const float* kb = (const float*)d_in[5];
  const float* vw = (const float*)d_in[6];
  const float* vb = (const float*)d_in[7];
  const float* ow = (const float*)d_in[8];
  const float* ob = (const float*)d_in[9];
  const float* c1w = (const float*)d_in[10];
  const float* c2w = (const float*)d_in[11];
  const float* norm_w = (const float*)d_in[12];
  const float* norm_b = (const float*)d_in[13];
  const float* proj_w = (const float*)d_in[14];
  const float* proj_b = (const float*)d_in[15];

  const size_t XSZ = (size_t)B_ * L_ * DM_;       // 8,388,608 elements
  const size_t FFSZ = (size_t)B_ * L_ * DFF_;     // 33,554,432 elements
  const int WQ = NL_ * DM_ * DM_;                 // 196,608 per Q/K/V/O array
  const int WC = NL_ * DFF_ * DM_;                // 786,432 per c1w/c2w array

  // fp32 region
  float* ws = (float*)d_ws;
  float* x    = ws;                    // activation (fp32)
  float* t1   = x  + XSZ;              // GEMM fp32 outputs / xh
  float* u    = t1 + XSZ;              // decomp1 output / final gelu
  float* v    = u  + XSZ;              // V projection (fp32)
  float* corr = v  + XSZ;
  float* wv   = corr + (size_t)B_ * L_;
  int*   dv   = (int*)(wv + B_ * TOPK_);
  // bf16 region (16B aligned)
  ushort_t* us = (ushort_t*)(((uintptr_t)(dv + B_ * TOPK_) + 15) & ~(uintptr_t)15);
  ushort_t* xb   = us;                 // bf16 copy of activation
  ushort_t* q16  = xb  + XSZ;
  ushort_t* k16  = q16 + XSZ;
  ushort_t* a16  = k16 + XSZ;          // attention output (bf16)
  ushort_t* u16  = a16 + XSZ;          // decomp1 output (bf16)
  ushort_t* ff16 = u16 + XSZ;          // FFN hidden (bf16)
  ushort_t* qwb  = ff16 + FFSZ;
  ushort_t* kwb  = qwb + WQ;
  ushort_t* vwb  = kwb + WQ;
  ushort_t* owb  = vwb + WQ;
  ushort_t* c1b  = owb + WQ;
  ushort_t* c2b  = c1b + WC;

  const int M = B_ * L_;                          // 32768
  const dim3 blk(256);
  const dim3 gD(M / BM_, DM_ / BN_);              // 256 x 4
  const dim3 gF1(M / BM_, DFF_ / BN_);            // 256 x 16
  const dim3 gCorr(L_ / BM_, L_ / BN_, B_);       // 8 x 16 x 32

  // one-time weight conversion (bandwidth-trivial)
  cvt_bf16_kernel<<<(WQ + 255) / 256, blk, 0, stream>>>(qw,  qwb, WQ);
  cvt_bf16_kernel<<<(WQ + 255) / 256, blk, 0, stream>>>(kw,  kwb, WQ);
  cvt_bf16_kernel<<<(WQ + 255) / 256, blk, 0, stream>>>(vw,  vwb, WQ);
  cvt_bf16_kernel<<<(WQ + 255) / 256, blk, 0, stream>>>(ow,  owb, WQ);
  cvt_bf16_kernel<<<(WC + 255) / 256, blk, 0, stream>>>(c1w, c1b, WC);
  cvt_bf16_kernel<<<(WC + 255) / 256, blk, 0, stream>>>(c2w, c2b, WC);

  token_embed_kernel<<<M, blk, 0, stream>>>(x_enc, token_w, x, xb);

  for (int l = 0; l < NL_; ++l) {
    const ushort_t* qwl = qwb + (size_t)l * DM_ * DM_;
    const ushort_t* kwl = kwb + (size_t)l * DM_ * DM_;
    const ushort_t* vwl = vwb + (size_t)l * DM_ * DM_;
    const ushort_t* owl = owb + (size_t)l * DM_ * DM_;
    const ushort_t* c1l = c1b + (size_t)l * DFF_ * DM_;
    const ushort_t* c2l = c2b + (size_t)l * DM_ * DFF_;
    const float* qbl = qb + l * DM_;
    const float* kbl = kb + l * DM_;
    const float* vbl = vb + l * DM_;
    const float* obl = ob + l * DM_;

    // Q/K bf16 out (feed corr GEMM), V fp32 out (feeds aggregate)
    gemm_bf16_wmma<false, true ><<<gD, blk, 0, stream>>>(xb, qwl, qbl, q16, DM_, DM_);
    gemm_bf16_wmma<false, true ><<<gD, blk, 0, stream>>>(xb, kwl, kbl, k16, DM_, DM_);
    gemm_bf16_wmma<false, false><<<gD, blk, 0, stream>>>(xb, vwl, vbl, v,   DM_, DM_);

    (void)hipMemsetAsync(corr, 0, (size_t)B_ * L_ * sizeof(float), stream);
    corr_wmma_kernel<<<gCorr, blk, 0, stream>>>(q16, k16, corr);
    topk_softmax_kernel<<<B_, blk, 0, stream>>>(corr, wv, dv);
    aggregate_kernel<<<M, blk, 0, stream>>>(v, wv, dv, a16);

    // O-projection (fp32 out) + residual + decomp (fp32 + bf16 out)
    gemm_bf16_wmma<false, false><<<gD, blk, 0, stream>>>(a16, owl, obl, t1, DM_, DM_);
    add_decomp_kernel<<<M, blk, 0, stream>>>(x, t1, u, u16);

    // FFN: GELU fused, hidden kept bf16; second GEMM fp32 out; decomp
    gemm_bf16_wmma<true , true ><<<gF1, blk, 0, stream>>>(u16, c1l, nullptr, ff16, DFF_, DM_);
    gemm_bf16_wmma<false, false><<<gD,  blk, 0, stream>>>(ff16, c2l, nullptr, t1, DM_, DFF_);
    add_decomp_kernel<<<M, blk, 0, stream>>>(u, t1, x, xb);
  }

  layernorm_kernel<<<M, blk, 0, stream>>>(x, norm_w, norm_b, t1);
  colmean_gelu_kernel<<<dim3(DM_, B_), blk, 0, stream>>>(t1, u);
  proj_kernel<<<dim3(3, B_), blk, 0, stream>>>(u, proj_w, proj_b, (float*)d_out);
}
